// Qwen3SparseMoeBlock_90134183674315
// MI455X (gfx1250) — compile-verified, matrix operand order checked by reading, not run on hardware
//
#include <hip/hip_runtime.h>
#include <hip/hip_bf16.h>

typedef __attribute__((ext_vector_type(16))) _Float16 v16h;
typedef __attribute__((ext_vector_type(8)))  _Float16 v8h;
typedef __attribute__((ext_vector_type(4)))  _Float16 v4h;
typedef __attribute__((ext_vector_type(8)))  float    v8f;

union ABFrag { v16h v; v8h h[2]; };

#define T_TOK 1024
#define H_DIM 1024
#define I_DIM 768
#define N_EXP 8
#define I_PAD 776    // 768 + 8 halfs (16B) pad to soften LDS bank conflicts

// per-wave LDS panel strips (double-buffered, wave-private => no block barrier)
#define LDSA_T    6144     // one tensor strip: 96 rows * 64 B
#define LDSA_BUF  12288    // gate + up strips
#define LDSA_WAVE 24576    // 2 buffers
#define LDSB_BUF  8192     // 128 rows * 64 B
#define LDSB_WAVE 16384    // 2 buffers
#define PANEL_LDS (8 * LDSA_WAVE)   // 196608 B

__device__ __forceinline__ unsigned ldsAddrOf(const void* p) {
    // generic shared pointer: low 32 bits are the LDS byte address
    return (unsigned)(uintptr_t)p;
}

__device__ __forceinline__ v8f wmma_f16(const ABFrag& a, const ABFrag& b, v8f c) {
    return __builtin_amdgcn_wmma_f32_16x16x32_f16(false, a.v, false, b.v,
                                                  (short)0, c, false, false);
}

// one async 16B-per-lane copy: LDS[vdst] = MEM[saddr + vaddr]
__device__ __forceinline__ void async_cp16(unsigned ldsAddr, unsigned gOff,
                                           const _Float16* sbase) {
    asm volatile("global_load_async_to_lds_b128 %0, %1, %2"
                 :: "v"(ldsAddr), "v"(gOff), "s"(sbase) : "memory");
}

// ---------------------------------------------------------------------------
// 0) zero output + expert counters
// ---------------------------------------------------------------------------
__global__ void zero_kernel(float4* __restrict__ out4, int n4,
                            int* __restrict__ counts, int ncnt) {
    int i = blockIdx.x * blockDim.x + threadIdx.x;
    if (i < n4) out4[i] = make_float4(0.f, 0.f, 0.f, 0.f);
    if (i < ncnt) counts[i] = 0;
}

// ---------------------------------------------------------------------------
// 1) fp32 -> f16 activation convert (vectorized)
// ---------------------------------------------------------------------------
__global__ void cvt_x_kernel(const float4* __restrict__ in4,
                             _Float16* __restrict__ out, int n4) {
    int i = blockIdx.x * blockDim.x + threadIdx.x;
    if (i < n4) {
        float4 v = in4[i];
        v4h h;
        h.x = (_Float16)v.x; h.y = (_Float16)v.y;
        h.z = (_Float16)v.z; h.w = (_Float16)v.w;
        *(v4h*)(out + (size_t)i * 4) = h;
    }
}

// ---------------------------------------------------------------------------
// 2) batched fp32 [B,R,C] -> f16 [B,C,R] tiled transpose (R,C multiples of 32)
// ---------------------------------------------------------------------------
__global__ void transpose_cvt_kernel(const float* __restrict__ in,
                                     _Float16* __restrict__ out,
                                     int R, int C) {
    __shared__ float tile[32][33];
    const float* src = in + (size_t)blockIdx.z * R * C;
    _Float16*    dst = out + (size_t)blockIdx.z * R * C;
    int c0 = blockIdx.x * 32;
    int r0 = blockIdx.y * 32;
    int tx = threadIdx.x;   // 0..31
    int ty = threadIdx.y;   // 0..7
#pragma unroll
    for (int i = 0; i < 32; i += 8)
        tile[ty + i][tx] = src[(size_t)(r0 + ty + i) * C + (c0 + tx)];
    __syncthreads();
#pragma unroll
    for (int i = 0; i < 32; i += 8)
        dst[(size_t)(c0 + ty + i) * R + (r0 + tx)] = (_Float16)tile[tx][ty + i];
}

// ---------------------------------------------------------------------------
// 3) router: one wave32 per token; top-2 of logits (softmax denominator
//    cancels under top-k renorm). Builds per-expert token lists.
// ---------------------------------------------------------------------------
__global__ void router_kernel(const float* __restrict__ x,
                              const float* __restrict__ gw,   // [H, E]
                              float* __restrict__ cw,         // [T, E]
                              int* __restrict__ counts,       // [E]
                              int* __restrict__ lists) {      // [E, T]
    int wave = threadIdx.x >> 5;
    int lane = threadIdx.x & 31;
    int t = blockIdx.x * 8 + wave;

    float acc[N_EXP];
#pragma unroll
    for (int e = 0; e < N_EXP; ++e) acc[e] = 0.f;

    const float* xr = x + (size_t)t * H_DIM;
    for (int h = lane; h < H_DIM; h += 32) {
        float xv = xr[h];
        const float* g = gw + (size_t)h * N_EXP;
#pragma unroll
        for (int e = 0; e < N_EXP; ++e) acc[e] += xv * g[e];
    }
#pragma unroll
    for (int e = 0; e < N_EXP; ++e)
#pragma unroll
        for (int off = 16; off > 0; off >>= 1)
            acc[e] += __shfl_xor(acc[e], off, 32);

    if (lane == 0) {
        int i1 = 0;
#pragma unroll
        for (int e = 1; e < N_EXP; ++e) if (acc[e] > acc[i1]) i1 = e;
        int i2 = (i1 == 0) ? 1 : 0;
#pragma unroll
        for (int e = 0; e < N_EXP; ++e)
            if (e != i1 && acc[e] > acc[i2]) i2 = e;
        float w2 = __expf(acc[i2] - acc[i1]);   // w1 = exp(0) = 1
        float inv = 1.0f / (1.0f + w2);
        cw[t * N_EXP + i1] = inv;
        cw[t * N_EXP + i2] = w2 * inv;
        int p1 = atomicAdd(&counts[i1], 1);
        lists[i1 * T_TOK + p1] = t;
        int p2 = atomicAdd(&counts[i2], 1);
        lists[i2 * T_TOK + p2] = t;
    }
}

// ---------------------------------------------------------------------------
// 4) routed expert MLP: one block = (expert, 16-token tile), 8 waves.
//    Weight K-panels are prefetched into wave-private LDS strips with
//    global_load_async_to_lds_b128 (double buffered on ASYNCcnt), WMMA
//    fragments then come from ds_load_b128. A-fragments are register
//    double-buffered one K-step ahead.
// ---------------------------------------------------------------------------
__global__ __launch_bounds__(256, 1)
void moe_expert_kernel(const _Float16* __restrict__ xh,    // [T, H]
                       const _Float16* __restrict__ wgT,   // [E, I, H]
                       const _Float16* __restrict__ wuT,   // [E, I, H]
                       const _Float16* __restrict__ wdT,   // [E, H, I]
                       const float* __restrict__ cw,       // [T, E]
                       const int* __restrict__ counts,     // [E]
                       const int* __restrict__ lists,      // [E, T]
                       float* __restrict__ out) {          // [T, H]
    const int tilesPerExpert = T_TOK / 16;
    int e    = blockIdx.x / tilesPerExpert;
    int tile = blockIdx.x % tilesPerExpert;
    int cnt  = counts[e];
    int row0 = tile * 16;
    if (row0 >= cnt) return;

    __shared__ char     panelLds[PANEL_LDS];
    __shared__ _Float16 actLds[16][I_PAD];
    __shared__ int      tokLds[16];
    __shared__ float    cwLds[16];

    if (threadIdx.x < 16) {
        int r = row0 + (int)threadIdx.x;
        int valid = (r < cnt);
        int tok = lists[e * T_TOK + (valid ? r : (cnt - 1))];
        tokLds[threadIdx.x] = tok;
        cwLds[threadIdx.x]  = valid ? cw[tok * N_EXP + e] : 0.0f;
    }
    __syncthreads();

    int lane = threadIdx.x & 31;
    int wave = threadIdx.x >> 5;
    int hi   = lane >> 4;       // lane group: K+0 vs K+8 ; C/D rows +8
    int l16  = lane & 15;       // A row / B,C,D column within a tile
    int kb   = hi * 8;

    int myTok = tokLds[l16];
    const _Float16* aBase = xh + (size_t)myTok * H_DIM + kb;
    const _Float16* wgE = wgT + (size_t)e * I_DIM * H_DIM;
    const _Float16* wuE = wuT + (size_t)e * I_DIM * H_DIM;
    int nb0 = wave * 96;

    // ---- Stage A: gate & up projections ----------------------------------
    // per-lane async-copy patterns: 12 x 16B chunks cover 96 rows x 64 B
    unsigned ldsA = ldsAddrOf(panelLds) + (unsigned)wave * LDSA_WAVE;
    unsigned gOffA[12], ldsOffA[12];
#pragma unroll
    for (int j = 0; j < 12; ++j) {
        int chunk = j * 32 + lane;
        int rowL  = chunk >> 2;
        int kc    = chunk & 3;
        gOffA[j]  = (unsigned)(((nb0 + rowL) * H_DIM + kc * 8) * 2);
        ldsOffA[j] = ldsA + (unsigned)(rowL * 64 + kc * 16);
    }
    auto issueA = [&](int k0n) {
        unsigned bo = ((unsigned)(k0n >> 5) & 1u) * LDSA_BUF;
        const _Float16* sg = wgE + k0n;
        const _Float16* su = wuE + k0n;
#pragma unroll
        for (int j = 0; j < 12; ++j)
            async_cp16(ldsOffA[j] + bo, gOffA[j], sg);
#pragma unroll
        for (int j = 0; j < 12; ++j)
            async_cp16(ldsOffA[j] + bo + LDSA_T, gOffA[j], su);
    };

    v8f accG[6], accU[6];
    v8f zero = {};
#pragma unroll
    for (int nt = 0; nt < 6; ++nt) { accG[nt] = zero; accU[nt] = zero; }

    issueA(0);
    ABFrag aCur, aNxt;
    aCur.h[0] = *(const v8h*)(aBase);
    aCur.h[1] = *(const v8h*)(aBase + 16);

#pragma unroll 1
    for (int k0 = 0; k0 < H_DIM; k0 += 32) {
        int k0n = k0 + 32;
        if (k0n < H_DIM) {
            issueA(k0n);                      // prefetch next panel
            aNxt.h[0] = *(const v8h*)(aBase + k0n);
            aNxt.h[1] = *(const v8h*)(aBase + k0n + 16);
            // 48 outstanding -> wait down to the 24 just issued
            asm volatile("s_wait_asynccnt 0x18" ::: "memory");
        } else {
            asm volatile("s_wait_asynccnt 0x0" ::: "memory");
        }
        const char* pb = panelLds + wave * LDSA_WAVE
                       + (((unsigned)(k0 >> 5) & 1u) * LDSA_BUF) + hi * 16;
#pragma unroll
        for (int nt = 0; nt < 6; ++nt) {
            const char* prow = pb + (nt * 16 + l16) * 64;
            ABFrag bg, bu;
            bg.h[0] = *(const v8h*)(prow);
            bg.h[1] = *(const v8h*)(prow + 32);
            bu.h[0] = *(const v8h*)(prow + LDSA_T);
            bu.h[1] = *(const v8h*)(prow + LDSA_T + 32);
            accG[nt] = wmma_f16(aCur, bg, accG[nt]);
            accU[nt] = wmma_f16(aCur, bu, accU[nt]);
        }
        aCur = aNxt;
    }

    // silu(g) * u -> f16 LDS tile (row-major [16][I])
#pragma unroll
    for (int nt = 0; nt < 6; ++nt) {
        int n = nb0 + nt * 16 + l16;
#pragma unroll
        for (int r = 0; r < 8; ++r) {
            int m = hi * 8 + r;
            float g = accG[nt][r];
            float u = accU[nt][r];
            float s = g / (1.0f + __expf(-g));
            actLds[m][n] = (_Float16)(s * u);
        }
    }
    __syncthreads();   // actLds ready; stage-A panel strips dead block-wide

    // ---- Stage B: down projection ----------------------------------------
    const _Float16* wdE  = wdT + (size_t)e * H_DIM * I_DIM;
    const _Float16* aLds = &actLds[l16][kb];
    int ob0 = wave * 128;

    unsigned ldsB = ldsAddrOf(panelLds) + (unsigned)wave * LDSB_WAVE;
    unsigned gOffB[16], ldsOffB[16];
#pragma unroll
    for (int j = 0; j < 16; ++j) {
        int chunk = j * 32 + lane;
        int rowL  = chunk >> 2;
        int kc    = chunk & 3;
        gOffB[j]  = (unsigned)(((ob0 + rowL) * I_DIM + kc * 8) * 2);
        ldsOffB[j] = ldsB + (unsigned)(rowL * 64 + kc * 16);
    }
    auto issueB = [&](int k0n) {
        unsigned bo = ((unsigned)(k0n >> 5) & 1u) * LDSB_BUF;
        const _Float16* sd = wdE + k0n;
#pragma unroll
        for (int j = 0; j < 16; ++j)
            async_cp16(ldsOffB[j] + bo, gOffB[j], sd);
    };

    v8f accO[8];
#pragma unroll
    for (int nt = 0; nt < 8; ++nt) accO[nt] = zero;

    issueB(0);
#pragma unroll 1
    for (int k0 = 0; k0 < I_DIM; k0 += 32) {
        int k0n = k0 + 32;
        if (k0n < I_DIM) {
            issueB(k0n);
            asm volatile("s_wait_asynccnt 0x10" ::: "memory");
        } else {
            asm volatile("s_wait_asynccnt 0x0" ::: "memory");
        }
        ABFrag a;
        a.h[0] = *(const v8h*)(aLds + k0);
        a.h[1] = *(const v8h*)(aLds + k0 + 16);
        const char* pb = panelLds + wave * LDSB_WAVE
                       + (((unsigned)(k0 >> 5) & 1u) * LDSB_BUF) + hi * 16;
#pragma unroll
        for (int nt = 0; nt < 8; ++nt) {
            const char* prow = pb + (nt * 16 + l16) * 64;
            ABFrag b;
            b.h[0] = *(const v8h*)(prow);
            b.h[1] = *(const v8h*)(prow + 32);
            accO[nt] = wmma_f16(a, b, accO[nt]);
        }
    }

    // ---- epilogue: scale by combine weight, atomically accumulate --------
#pragma unroll
    for (int r = 0; r < 8; ++r) {
        int m = hi * 8 + r;
        float w = cwLds[m];
        if (w == 0.0f) continue;         // pad rows
        float* orow = out + (size_t)tokLds[m] * H_DIM;
#pragma unroll
        for (int nt = 0; nt < 8; ++nt) {
            int n = ob0 + nt * 16 + l16;
            unsafeAtomicAdd(&orow[n], accO[nt][r] * w);
        }
    }
}

// ---------------------------------------------------------------------------
extern "C" void kernel_launch(void* const* d_in, const int* in_sizes, int n_in,
                              void* d_out, int out_size, void* d_ws, size_t ws_size,
                              hipStream_t stream) {
    const float* x      = (const float*)d_in[0];  // [2,512,1024]
    const float* gate_w = (const float*)d_in[1];  // [1024,8]
    const float* w_gate = (const float*)d_in[2];  // [8,1024,768]
    const float* w_up   = (const float*)d_in[3];  // [8,1024,768]
    const float* w_down = (const float*)d_in[4];  // [8,768,1024]
    float* out = (float*)d_out;

    char* ws = (char*)d_ws;
    size_t off = 0;
    auto alloc = [&](size_t bytes) -> void* {
        off = (off + 255) & ~(size_t)255;
        void* p = ws + off;
        off += bytes;
        return p;
    };
    _Float16* xh  = (_Float16*)alloc((size_t)T_TOK * H_DIM * 2);
    _Float16* wgT = (_Float16*)alloc((size_t)N_EXP * I_DIM * H_DIM * 2);
    _Float16* wuT = (_Float16*)alloc((size_t)N_EXP * I_DIM * H_DIM * 2);
    _Float16* wdT = (_Float16*)alloc((size_t)N_EXP * H_DIM * I_DIM * 2);
    float* cw     = (float*)alloc((size_t)T_TOK * N_EXP * 4);
    int* counts   = (int*)alloc((size_t)N_EXP * 4);
    int* lists    = (int*)alloc((size_t)N_EXP * T_TOK * 4);

    int n4 = T_TOK * H_DIM / 4;
    zero_kernel<<<(n4 + 255) / 256, 256, 0, stream>>>((float4*)out, n4, counts, N_EXP);
    cvt_x_kernel<<<(n4 + 255) / 256, 256, 0, stream>>>((const float4*)x, xh, n4);

    dim3 tb(32, 8);
    transpose_cvt_kernel<<<dim3(I_DIM / 32, H_DIM / 32, N_EXP), tb, 0, stream>>>(
        w_gate, wgT, H_DIM, I_DIM);   // [E,H,I] -> [E,I,H]
    transpose_cvt_kernel<<<dim3(I_DIM / 32, H_DIM / 32, N_EXP), tb, 0, stream>>>(
        w_up, wuT, H_DIM, I_DIM);     // [E,H,I] -> [E,I,H]
    transpose_cvt_kernel<<<dim3(H_DIM / 32, I_DIM / 32, N_EXP), tb, 0, stream>>>(
        w_down, wdT, I_DIM, H_DIM);   // [E,I,H] -> [E,H,I]

    router_kernel<<<T_TOK / 8, 256, 0, stream>>>(x, gate_w, cw, counts, lists);

    moe_expert_kernel<<<N_EXP * (T_TOK / 16), 256, 0, stream>>>(
        xh, wgT, wuT, wdT, cw, counts, lists, out);
}
